// NeuralCDE_26233660244178
// MI455X (gfx1250) — matrix operand order repaired and checked by reading, not compiled, not run on hardware
//
#include <hip/hip_runtime.h>
#include <hip/hip_bf16.h>

typedef __attribute__((ext_vector_type(16))) _Float16 v16h;
typedef __attribute__((ext_vector_type(8)))  _Float16 v8h;
typedef __attribute__((ext_vector_type(8)))  float    v8f;
typedef __attribute__((ext_vector_type(4)))  float    v4f;
typedef __attribute__((ext_vector_type(8)))  unsigned v8u;

#define NSTEPS 255
#define CIN    6
#define HDIM   32
#define MIDD   128
#define OUTD   192   // HDIM * CIN

__device__ __forceinline__ unsigned pack2(float a, float b) {
    auto t = __builtin_amdgcn_cvt_pkrtz(a, b);   // __fp16x2, low = a, high = b
    return __builtin_bit_cast(unsigned, t);
}

__device__ __forceinline__ float fast_tanh(float x) {
#if __has_builtin(__builtin_amdgcn_tanhf)
    return __builtin_amdgcn_tanhf(x);            // v_tanh_f32 (gfx1250 trans op)
#else
    float e = __expf(2.0f * x);
    return (e - 1.0f) * __builtin_amdgcn_rcpf(e + 1.0f);
#endif
}

// Load a 16x32 f16 A-fragment for one wave from LDS stored row-major.
// Lane l<16: M=l, K = 0..15; lane l>=16: M=l-16, K = 16..31.
__device__ __forceinline__ v16h load_afrag(const _Float16* p, int off0, int off1) {
    v8h lo = *(const v8h*)(p + off0);
    v8h hi = *(const v8h*)(p + off1);
    return __builtin_shufflevector(lo, hi, 0,1,2,3,4,5,6,7,8,9,10,11,12,13,14,15);
}

// C/D-layout f32 fragment of a per-M (broadcast over N) bias vector.
__device__ __forceinline__ v8f load_cfrag(const float* p, int off) {
    v4f a = *(const v4f*)(p + off);
    v4f b = *(const v4f*)(p + off + 4);
    return __builtin_shufflevector(a, b, 0,1,2,3,4,5,6,7);
}

// Combine two packed-f16 C/D tiles (K-halves) into one 32xK B fragment.
// pkLo = tile covering K 0..15, pkHi = tile covering K 16..31 (packed pairs).
// Target: b[r]   = [lanes<16: pkLo(l)   | lanes>=16: pkHi(l-16)]
//         b[r+4] = [lanes<16: pkLo(l+16)| lanes>=16: pkHi(l)   ]
__device__ __forceinline__ v16h combineB(const unsigned pkLo[4], const unsigned pkHi[4], bool hh) {
    v8u b;
    #pragma unroll
    for (int r = 0; r < 4; ++r) {
#if __has_builtin(__builtin_amdgcn_permlane16_swap)
        // v_permlane16_swap_b32: swaps lanes 16-31 of arg0 with lanes 0-15 of arg1.
        auto res = __builtin_amdgcn_permlane16_swap(pkLo[r], pkHi[r], false, false);
        b[r]     = res[0];
        b[r + 4] = res[1];
#else
        unsigned Y = hh ? pkHi[r] : pkLo[r];
        unsigned Z = hh ? pkLo[r] : pkHi[r];
        unsigned W = __shfl_xor(Z, 16);
        b[r]     = hh ? W : Y;
        b[r + 4] = hh ? Y : W;
#endif
    }
    return __builtin_bit_cast(v16h, b);
}

// vf(z, dx):  h = tanh(z@W1 + b1);  g' = h@W2' + b2';  y = sum_i g'[:, i*32+h] * dx_i
// Transposed mapping: weights are WMMA A (from LDS), activations are B (registers).
// z/k/y live as two C/D-layout f32 tiles (m = hidden 0..15 / 16..31, n = batch).
__device__ __forceinline__ void vf(const _Float16* sW1, const _Float16* sW2,
                                   const float* sB1, const float* sB2,
                                   int n, int hh, bool hiHalf,
                                   v8f z0, v8f z1, const float* dx,
                                   v8f& y0, v8f& y1)
{
    // ---- z (2 C/D tiles, f32) -> one B fragment (K=32 hidden, N=16 batch, f16)
    unsigned pk0[4], pk1[4];
    #pragma unroll
    for (int r = 0; r < 4; ++r) {
        pk0[r] = pack2(z0[2*r], z0[2*r+1]);
        pk1[r] = pack2(z1[2*r], z1[2*r+1]);
    }
    v16h Bz = combineB(pk0, pk1, hiHalf);

    // ---- stage 1: h^T = tanh(W1^T(128x32) x z^T(32x16) + b1); convert the 8
    // output tiles pairwise into 4 B fragments along K=MID.
    v16h Bh[4];
    #pragma unroll
    for (int t = 0; t < 4; ++t) {
        int mt0 = 2*t, mt1 = 2*t + 1;
        v16h A0 = load_afrag(sW1, (mt0*16 + n)*HDIM + hh*16, (mt0*16 + n)*HDIM + 8 + hh*16);
        v16h A1 = load_afrag(sW1, (mt1*16 + n)*HDIM + hh*16, (mt1*16 + n)*HDIM + 8 + hh*16);
        v8f  C0 = load_cfrag(sB1, mt0*16 + hh*8);
        v8f  C1 = load_cfrag(sB1, mt1*16 + hh*8);
        v8f hc0 = __builtin_amdgcn_wmma_f32_16x16x32_f16(false, A0, false, Bz, (short)0, C0, false, false);
        v8f hc1 = __builtin_amdgcn_wmma_f32_16x16x32_f16(false, A1, false, Bz, (short)0, C1, false, false);
        #pragma unroll
        for (int r = 0; r < 8; ++r) { hc0[r] = fast_tanh(hc0[r]); hc1[r] = fast_tanh(hc1[r]); }
        unsigned q0[4], q1[4];
        #pragma unroll
        for (int r = 0; r < 4; ++r) {
            q0[r] = pack2(hc0[2*r], hc0[2*r+1]);
            q1[r] = pack2(hc1[2*r], hc1[2*r+1]);
        }
        Bh[t] = combineB(q0, q1, hiHalf);
    }

    // ---- stage 2 + contraction: 12 output tiles (c' = i*32 + h), each = 4 WMMAs
    // over K=MID; tile t contributes dx[t/2] to y-tile (t&1).
    y0 = v8f{}; y1 = v8f{};
    #pragma unroll
    for (int t = 0; t < 12; ++t) {
        v8f g = load_cfrag(sB2, t*16 + hh*8);
        #pragma unroll
        for (int kf = 0; kf < 4; ++kf) {
            v16h A = load_afrag(sW2, (t*16 + n)*MIDD + kf*32 + hh*16,
                                      (t*16 + n)*MIDD + kf*32 + 8 + hh*16);
            g = __builtin_amdgcn_wmma_f32_16x16x32_f16(false, A, false, Bh[kf], (short)0, g, false, false);
        }
        float d = dx[t >> 1];
        if (t & 1) y1 += g * d; else y0 += g * d;
    }
}

__global__ void __launch_bounds__(32)
neural_cde_kernel(const float* __restrict__ times,
                  const float* __restrict__ cb, const float* __restrict__ cc,
                  const float* __restrict__ cd,
                  const float* __restrict__ W1, const float* __restrict__ b1,
                  const float* __restrict__ W2, const float* __restrict__ b2,
                  const float* __restrict__ Wl, const float* __restrict__ bl,
                  float* __restrict__ out)
{
    __shared__ __attribute__((aligned(16))) _Float16 sW1[MIDD * HDIM];   // W1^T (row=mid, col=hidden)
    __shared__ __attribute__((aligned(16))) _Float16 sW2[OUTD * MIDD];   // W2'^T (row=c'=i*32+h, col=mid)
    __shared__ __attribute__((aligned(16))) float    sB1[MIDD];
    __shared__ __attribute__((aligned(16))) float    sB2[OUTD];          // permuted b2
    __shared__ __attribute__((aligned(16))) float    sZ[16 * HDIM];      // final z staging

    const int tid = threadIdx.x;   // one wave per block

    // ---- cooperative LDS fill (one-time) ----
    for (int idx = tid; idx < MIDD * HDIM; idx += 32) {
        int m = idx >> 5, k = idx & 31;                 // W1^T[m][k] = W1[k][m]
        sW1[idx] = (_Float16)W1[k * MIDD + m];
    }
    for (int idx = tid; idx < OUTD * MIDD; idx += 32) {
        int row = idx >> 7, k = idx & 127;              // row = c' = i*32 + h
        int i = row >> 5, h = row & 31;
        sW2[idx] = (_Float16)W2[k * OUTD + (h * CIN + i)];
    }
    for (int idx = tid; idx < MIDD; idx += 32) sB1[idx] = b1[idx];
    for (int idx = tid; idx < OUTD; idx += 32) {
        int i = idx >> 5, h = idx & 31;
        sB2[idx] = b2[h * CIN + i];
    }
    __syncthreads();

    const int lane = tid & 31;
    const int n = lane & 15, hh = lane >> 4;
    const bool hiHalf = (hh != 0);
    const long bg = (long)blockIdx.x * 16 + n;

    const float* pb = cb + bg * (long)(NSTEPS * CIN);
    const float* pc = cc + bg * (long)(NSTEPS * CIN);
    const float* pd = cd + bg * (long)(NSTEPS * CIN);

    v8f z0 = v8f{}, z1 = v8f{};

    #pragma unroll 1
    for (int s = 0; s < NSTEPS; ++s) {
        const float dt = times[s + 1] - times[s];
        const long off = (long)s * CIN;
        float bco[6], cco[6], dco[6];
        #pragma unroll
        for (int i = 0; i < 6; i += 2) {
            float2 vb = *(const float2*)(pb + off + i);
            float2 vc = *(const float2*)(pc + off + i);
            float2 vd = *(const float2*)(pd + off + i);
            bco[i] = vb.x; bco[i+1] = vb.y;
            cco[i] = vc.x; cco[i+1] = vc.y;
            dco[i] = vd.x; dco[i+1] = vd.y;
        }
        const float f2 = 0.5f * dt;
        float dx1[6], dx2[6], dx4[6];
        #pragma unroll
        for (int i = 0; i < 6; ++i) {
            dx1[i] = bco[i];
            dx2[i] = bco[i] + cco[i] * f2 + dco[i] * (f2 * f2);
            dx4[i] = bco[i] + cco[i] * dt + dco[i] * (dt * dt);
        }

        v8f k1a, k1b, k2a, k2b, k3a, k3b, k4a, k4b;
        vf(sW1, sW2, sB1, sB2, n, hh, hiHalf, z0, z1, dx1, k1a, k1b);
        vf(sW1, sW2, sB1, sB2, n, hh, hiHalf, z0 + k1a * f2, z1 + k1b * f2, dx2, k2a, k2b);
        vf(sW1, sW2, sB1, sB2, n, hh, hiHalf, z0 + k2a * f2, z1 + k2b * f2, dx2, k3a, k3b);
        vf(sW1, sW2, sB1, sB2, n, hh, hiHalf, z0 + k3a * dt, z1 + k3b * dt, dx4, k4a, k4b);

        const float w = dt * (1.0f / 6.0f);
        z0 += (k1a + k2a * 2.0f + k3a * 2.0f + k4a) * w;
        z1 += (k1b + k2b * 2.0f + k3b * 2.0f + k4b) * w;
    }

    // ---- readout: out = z @ Wl + bl  (stage z through LDS once) ----
    {
        float* zp = &sZ[n * HDIM + hh * 8];
        #pragma unroll
        for (int r = 0; r < 8; ++r) { zp[r] = z0[r]; zp[16 + r] = z1[r]; }
    }
    __syncthreads();

    #pragma unroll
    for (int j = 0; j < 3; ++j) {
        int idx = lane * 3 + j;           // 96 outputs per wave = 16 rows * 6 ch
        int brow = idx / 6, ch = idx - brow * 6;
        const float* zr = &sZ[brow * HDIM];
        float acc = bl[ch];
        #pragma unroll
        for (int h = 0; h < HDIM; ++h) acc += zr[h] * Wl[h * CIN + ch];
        out[((long)blockIdx.x * 16 + brow) * CIN + ch] = acc;
    }
}

extern "C" void kernel_launch(void* const* d_in, const int* in_sizes, int n_in,
                              void* d_out, int out_size, void* d_ws, size_t ws_size,
                              hipStream_t stream) {
    const float* times = (const float*)d_in[0];
    // d_in[1] = coeff_a (unused by the reference vector field)
    const float* cb = (const float*)d_in[2];
    const float* cc = (const float*)d_in[3];
    const float* cd = (const float*)d_in[4];
    const float* W1 = (const float*)d_in[5];
    const float* b1 = (const float*)d_in[6];
    const float* W2 = (const float*)d_in[7];
    const float* b2 = (const float*)d_in[8];
    const float* Wl = (const float*)d_in[9];
    const float* bl = (const float*)d_in[10];
    float* out = (float*)d_out;

    const int Bn = in_sizes[2] / (NSTEPS * CIN);   // 8192
    const int blocks = Bn / 16;                    // 16 batch rows per 1-wave block -> 512 blocks
    hipLaunchKernelGGL(neural_cde_kernel, dim3(blocks), dim3(32), 0, stream,
                       times, cb, cc, cd, W1, b1, W2, b2, Wl, bl, out);
}
